// MultiHeadKoopmanModule_3590592659973
// MI455X (gfx1250) — compile-verified, matrix operand order checked by reading, not run on hardware
//
#include <hip/hip_runtime.h>

// ---------------------------------------------------------------------------
// MultiHeadKoopman for MI455X (gfx1250, wave32, WMMA)
//  - big GEMMs: v_wmma_f32_16x16x32_bf16 fed by double-buffered
//    global_load_async_to_lds_b128 (ASYNCcnt-tracked) LDS staging
//  - per-head 48x48 algebra: fp32 VALU, operator chain precomposed into a
//    single 64x48 matrix per (b,h,k)
// ---------------------------------------------------------------------------

typedef __attribute__((ext_vector_type(16))) __bf16 v16bf;
typedef __attribute__((ext_vector_type(8)))  __bf16 v8bf;
typedef __attribute__((ext_vector_type(8)))  float  v8f;

constexpr int Bz   = 4;
constexpr int T    = 2048;
constexpr int Dm   = 1024;
constexpr int H    = 16;
constexpr int HD   = 64;
constexpr int R    = 48;
constexpr int KOPS = 4;
constexpr int BT   = Bz * T;    // 8192
constexpr int HR   = H * R;     // 768
constexpr int HHD  = H * HD;    // 1024
constexpr float LN_EPS = 1e-5f;

// ---------------------------------------------------------------------------
// LayerNorm -> bf16
// ---------------------------------------------------------------------------
__global__ __launch_bounds__(256) void ln_kernel(const float* __restrict__ x,
                                                 const float* __restrict__ gam,
                                                 const float* __restrict__ bet,
                                                 __bf16* __restrict__ out)
{
  __shared__ float red[256];
  int row = blockIdx.x;
  const float* xr = x + (size_t)row * Dm;
  float s = 0.f, ss = 0.f;
  for (int i = threadIdx.x; i < Dm; i += 256) { float v = xr[i]; s += v; ss += v * v; }
  red[threadIdx.x] = s; __syncthreads();
  for (int o = 128; o > 0; o >>= 1) { if (threadIdx.x < o) red[threadIdx.x] += red[threadIdx.x + o]; __syncthreads(); }
  float mean = red[0] / (float)Dm;
  __syncthreads();
  red[threadIdx.x] = ss; __syncthreads();
  for (int o = 128; o > 0; o >>= 1) { if (threadIdx.x < o) red[threadIdx.x] += red[threadIdx.x + o]; __syncthreads(); }
  float var = red[0] / (float)Dm - mean * mean;
  float inv = rsqrtf(var + LN_EPS);
  __bf16* orow = out + (size_t)row * Dm;
  for (int i = threadIdx.x; i < Dm; i += 256)
    orow[i] = (__bf16)((xr[i] - mean) * inv * gam[i] + bet[i]);
}

// ---------------------------------------------------------------------------
// f32 (rows x cols) -> bf16 transposed (cols x rows)
// ---------------------------------------------------------------------------
__global__ void transpose_bf16_kernel(const float* __restrict__ in, __bf16* __restrict__ out,
                                      int rows, int cols)
{
  int i = blockIdx.x * 256 + threadIdx.x;
  if (i >= rows * cols) return;
  int r = i / cols, c = i % cols;
  out[(size_t)c * rows + r] = (__bf16)in[(size_t)r * cols + c];
}

__global__ void cvt_bf16_kernel(const float* __restrict__ in, __bf16* __restrict__ out, int n)
{
  int i = blockIdx.x * 256 + threadIdx.x;
  if (i < n) out[i] = (__bf16)in[i];
}

// ---------------------------------------------------------------------------
// bf16 WMMA GEMM, async-LDS double-buffered.
//   C(MxN,f32) = alpha * A(MxK,bf16 rowmajor) * Bt^T  (Bt is N x K rowmajor)
// Block: 256 threads = 8 waves; block tile 128(M) x 64(N); K-step 32.
// Per K-step each thread issues 3x global_load_async_to_lds_b128 (12KB/block)
// into the next buffer, waits s_wait_asynccnt, barriers, computes from LDS.
// Fragment layout per CDNA5 ISA (16-bit A 16x32 / B 32x16):
//   lane<16 -> K blocks {0,2}; lane>=16 -> K blocks {1,3}; 8 halves each.
// ---------------------------------------------------------------------------
__device__ __forceinline__ v16bf load_frag(const __bf16* p)
{
  v8bf lo = *reinterpret_cast<const v8bf*>(p);
  v8bf hi = *reinterpret_cast<const v8bf*>(p + 16);
  return __builtin_shufflevector(lo, hi, 0, 1, 2, 3, 4, 5, 6, 7, 8, 9, 10, 11, 12, 13, 14, 15);
}

__device__ __forceinline__ void async_cp16(unsigned lds_addr, const __bf16* gptr)
{
  unsigned long long ga = (unsigned long long)(uintptr_t)gptr;
  asm volatile("global_load_async_to_lds_b128 %0, %1, off"
               :: "v"(lds_addr), "v"(ga) : "memory");
}

__global__ __launch_bounds__(256) void gemm_bf16_wmma(const __bf16* __restrict__ A,
                                                      const __bf16* __restrict__ Bt,
                                                      float* __restrict__ C,
                                                      int M, int N, int K,
                                                      const float* __restrict__ alpha_ptr,
                                                      int alpha_sigmoid)
{
  __shared__ __bf16 shA[2][128 * 32];   // 8KB per buffer
  __shared__ __bf16 shB[2][64 * 32];    // 4KB per buffer

  int tid = threadIdx.x;
  int wave = tid >> 5;
  int lane = tid & 31;
  int lr = lane & 15;
  int hi = lane >> 4;
  int m0 = blockIdx.x * 128;
  int n0 = blockIdx.y * 64;

  // cooperative-copy mapping: each thread owns one 16B chunk per tile-slab
  int crow = tid >> 2;          // 0..63
  int cchk = (tid & 3) * 8;     // half-element offset of 16B chunk
  const __bf16* gA0 = A + (size_t)(m0 + crow) * K + cchk;
  const __bf16* gA1 = A + (size_t)(m0 + 64 + crow) * K + cchk;
  const __bf16* gB  = Bt + (size_t)(n0 + crow) * K + cchk;
  unsigned lA0 = (unsigned)(uintptr_t)&shA[0][crow * 32 + cchk];
  unsigned lA1 = (unsigned)(uintptr_t)&shA[0][(64 + crow) * 32 + cchk];
  unsigned lB  = (unsigned)(uintptr_t)&shB[0][crow * 32 + cchk];

  v8f c0 = {0.f, 0.f, 0.f, 0.f, 0.f, 0.f, 0.f, 0.f};
  v8f c1 = c0, c2 = c0, c3 = c0;

  const int nsteps = K >> 5;

  // prologue: stage K-step 0 into buffer 0
  async_cp16(lA0, gA0);
  async_cp16(lA1, gA1);
  async_cp16(lB,  gB);

  for (int it = 0; it < nsteps; ++it) {
    int cur = it & 1;
    if (it + 1 < nsteps) {
      int kb = (it + 1) << 5;
      unsigned boff = (cur ^ 1) ? 8192u : 0u;   // byte offset of next A buffer
      unsigned boffB = (cur ^ 1) ? 4096u : 0u;  // byte offset of next B buffer
      async_cp16(lA0 + boff, gA0 + kb);
      async_cp16(lA1 + boff, gA1 + kb);
      async_cp16(lB + boffB, gB + kb);
      asm volatile("s_wait_asynccnt 3" ::: "memory");   // older batch retired
    } else {
      asm volatile("s_wait_asynccnt 0" ::: "memory");
    }
    __syncthreads();   // cur buffers fully written by all waves

    const __bf16* Ab = &shA[cur][(wave * 16 + lr) * 32 + 8 * hi];
    const __bf16* Bb = &shB[cur][lr * 32 + 8 * hi];
    v16bf a  = load_frag(Ab);
    v16bf b0 = load_frag(Bb);
    v16bf b1 = load_frag(Bb + 16 * 32);
    v16bf b2 = load_frag(Bb + 32 * 32);
    v16bf b3 = load_frag(Bb + 48 * 32);
    c0 = __builtin_amdgcn_wmma_f32_16x16x32_bf16(false, a, false, b0, (short)0, c0, false, false);
    c1 = __builtin_amdgcn_wmma_f32_16x16x32_bf16(false, a, false, b1, (short)0, c1, false, false);
    c2 = __builtin_amdgcn_wmma_f32_16x16x32_bf16(false, a, false, b2, (short)0, c2, false, false);
    c3 = __builtin_amdgcn_wmma_f32_16x16x32_bf16(false, a, false, b3, (short)0, c3, false, false);

    __syncthreads();   // all waves done reading cur before it is overwritten
  }

  float alpha = 1.0f;
  if (alpha_ptr) {
    alpha = *alpha_ptr;
    if (alpha_sigmoid) alpha = 1.0f / (1.0f + expf(-alpha));
  }

  // C/D layout: VGPR r, lanes 0-15 -> M=r, lanes 16-31 -> M=r+8; N=lane&15
  size_t cbase = (size_t)(m0 + wave * 16 + 8 * hi) * N + n0 + lr;
#pragma unroll
  for (int r = 0; r < 8; r++) {
    float* cp = C + cbase + (size_t)r * N;
    cp[0]  = alpha * c0[r];
    cp[16] = alpha * c1[r];
    cp[32] = alpha * c2[r];
    cp[48] = alpha * c3[r];
  }
}

// ---------------------------------------------------------------------------
// G = pk^T pk, M = pk[1:]^T pk[:-1]  per (b,h); LDS-staged over prefix
// ---------------------------------------------------------------------------
__global__ __launch_bounds__(256) void gm_kernel(const float* __restrict__ keys,
                                                 float* __restrict__ G, float* __restrict__ Mo,
                                                 const int* __restrict__ plp)
{
  __shared__ float sk[33 * 48];
  int bh = blockIdx.x, b = bh >> 4, h = bh & 15, tid = threadIdx.x;
  int pl = *plp; if (pl < 1) pl = 1; if (pl > T - 1) pl = T - 1;
  const float* kb = keys + (size_t)b * T * HR + h * R;
  float ga[9], ma[9]; int rp[9], sp[9];
#pragma unroll
  for (int p = 0; p < 9; p++) { int idx = tid + 256 * p; rp[p] = idx / 48; sp[p] = idx % 48; ga[p] = 0.f; ma[p] = 0.f; }
  for (int l0 = 0; l0 < pl; l0 += 32) {
    int nrow = min(32, pl - l0);
    int ext = (l0 + nrow < pl) ? 1 : 0;
    int tot = (nrow + ext) * 48;
    __syncthreads();
    for (int t2 = tid; t2 < tot; t2 += 256) {
      int rr = t2 / 48, cc = t2 % 48;
      sk[rr * 48 + cc] = kb[(size_t)(l0 + rr) * HR + cc];
    }
    __syncthreads();
    for (int l = 0; l < nrow; l++) {
      bool dm = (l0 + l + 1 < pl);
#pragma unroll
      for (int p = 0; p < 9; p++) {
        float ks = sk[l * 48 + sp[p]];
        ga[p] += sk[l * 48 + rp[p]] * ks;
        if (dm) ma[p] += sk[(l + 1) * 48 + rp[p]] * ks;
      }
    }
  }
  for (int p = 0; p < 9; p++) {
    int idx = tid + 256 * p;
    G[(size_t)bh * 2304 + idx] = ga[p];
    Mo[(size_t)bh * 2304 + idx] = ma[p];
  }
}

// ---------------------------------------------------------------------------
// Cv[d][r] = sum_l pv[l][d] * pk[l][r]  (64 x 48 per (b,h))
// ---------------------------------------------------------------------------
__global__ __launch_bounds__(256) void cv_kernel(const float* __restrict__ keys,
                                                 const float* __restrict__ values,
                                                 float* __restrict__ Cv,
                                                 const int* __restrict__ plp)
{
  __shared__ float sk[32 * 48];
  __shared__ float sv[32 * 64];
  int bh = blockIdx.x, b = bh >> 4, h = bh & 15, tid = threadIdx.x;
  int pl = *plp; if (pl < 1) pl = 1; if (pl > T - 1) pl = T - 1;
  const float* kb = keys + (size_t)b * T * HR + h * R;
  const float* vb = values + (size_t)b * T * HHD + h * HD;
  float ca[12]; int dp[12], rp[12];
#pragma unroll
  for (int p = 0; p < 12; p++) { int idx = tid + 256 * p; dp[p] = idx / 48; rp[p] = idx % 48; ca[p] = 0.f; }
  for (int l0 = 0; l0 < pl; l0 += 32) {
    int nrow = min(32, pl - l0);
    __syncthreads();
    for (int t2 = tid; t2 < nrow * 48; t2 += 256) {
      int rr = t2 / 48, cc = t2 % 48;
      sk[rr * 48 + cc] = kb[(size_t)(l0 + rr) * HR + cc];
    }
    for (int t2 = tid; t2 < nrow * 64; t2 += 256) {
      int rr = t2 / 64, cc = t2 % 64;
      sv[rr * 64 + cc] = vb[(size_t)(l0 + rr) * HHD + cc];
    }
    __syncthreads();
    for (int l = 0; l < nrow; l++) {
#pragma unroll
      for (int p = 0; p < 12; p++)
        ca[p] += sv[l * 64 + dp[p]] * sk[l * 48 + rp[p]];
    }
  }
  for (int p = 0; p < 12; p++) Cv[(size_t)bh * 3072 + tid + 256 * p] = ca[p];
}

// ---------------------------------------------------------------------------
// Per (b,h): Cholesky(G + ridge I); A_w with spectral clamp (power iteration);
// B_v; then precompose Cmat = B_v * L * A_w^2 * L^-1  (64 x 48)
// ---------------------------------------------------------------------------
__global__ __launch_bounds__(64) void factor_kernel(const float* __restrict__ G,
                                                    const float* __restrict__ Mb,
                                                    const float* __restrict__ Cv,
                                                    float* __restrict__ Cm,
                                                    const float* __restrict__ log_ridges,
                                                    const float* __restrict__ log_gammas,
                                                    int k)
{
  __shared__ float Ls[2304];
  __shared__ float Aw[2304];
  __shared__ float Wa[2304];
  __shared__ float Wb[3072];
  __shared__ float vv[48];
  __shared__ float uu[48];
  __shared__ float snorm;
  int bh = blockIdx.x, tid = threadIdx.x;
  float ridge = expf(log_ridges[k]);
  float gam = expf(log_gammas[k]);

  for (int i = tid; i < 2304; i += 64) {
    float v = G[(size_t)bh * 2304 + i];
    if (i / 48 == i % 48) v += ridge;
    Ls[i] = v;
  }
  __syncthreads();

  // in-place lower Cholesky
  for (int j = 0; j < 48; j++) {
    if (tid == 0) Ls[j * 48 + j] = sqrtf(fmaxf(Ls[j * 48 + j], 1e-30f));
    __syncthreads();
    float dj = Ls[j * 48 + j];
    if (tid > j && tid < 48) Ls[tid * 48 + j] /= dj;
    __syncthreads();
    if (tid > j && tid < 48) {
      float lij = Ls[tid * 48 + j];
      for (int c = j + 1; c <= tid; c++) Ls[tid * 48 + c] -= lij * Ls[c * 48 + j];
    }
    __syncthreads();
  }

  // U = L^-1 M -> Wa
  if (tid < 48) {
    float x[48];
    for (int i = 0; i < 48; i++) {
      float a = Mb[(size_t)bh * 2304 + i * 48 + tid];
      for (int t2 = 0; t2 < i; t2++) a -= Ls[i * 48 + t2] * x[t2];
      x[i] = a / Ls[i * 48 + i];
      Wa[i * 48 + tid] = x[i];
    }
  }
  __syncthreads();

  // A_w = (L^-1 U^T)^T : row tid of A_w
  if (tid < 48) {
    float x[48];
    for (int i = 0; i < 48; i++) {
      float a = Wa[tid * 48 + i];
      for (int t2 = 0; t2 < i; t2++) a -= Ls[i * 48 + t2] * x[t2];
      x[i] = a / Ls[i * 48 + i];
    }
    for (int i = 0; i < 48; i++) Aw[tid * 48 + i] = x[i];
  }
  __syncthreads();

  // power iteration: sigma_max(A_w) = sqrt(lambda_max(A^T A))
  if (tid < 48) vv[tid] = 0.14433757f;
  __syncthreads();
  for (int it = 0; it < 24; it++) {
    if (tid < 48) { float a = 0.f; for (int j = 0; j < 48; j++) a += Aw[tid * 48 + j] * vv[j]; uu[tid] = a; }
    __syncthreads();
    if (tid < 48) { float a = 0.f; for (int i = 0; i < 48; i++) a += Aw[i * 48 + tid] * uu[i]; vv[tid] = a; }
    __syncthreads();
    if (tid == 0) { float n = 0.f; for (int i = 0; i < 48; i++) n += vv[i] * vv[i]; snorm = sqrtf(n); }
    __syncthreads();
    if (tid < 48) vv[tid] /= fmaxf(snorm, 1e-30f);
    __syncthreads();
  }
  float sigma = fmaxf(sqrtf(fmaxf(snorm, 0.f)), 1e-8f);
  float scale = fminf(gam, 1.f) / fmaxf(sigma, 1.f);
  for (int i = tid; i < 2304; i += 64) Aw[i] *= scale;
  __syncthreads();

  // B_v (64x48) -> Wb : Y = L^-1 Cv^T ; Bv^T = L^-T Y
  {
    float x[48], y[48];
    for (int i = 0; i < 48; i++) {
      float a = Cv[(size_t)bh * 3072 + tid * 48 + i];
      for (int t2 = 0; t2 < i; t2++) a -= Ls[i * 48 + t2] * x[t2];
      x[i] = a / Ls[i * 48 + i];
    }
    for (int i = 47; i >= 0; i--) {
      float a = x[i];
      for (int t2 = i + 1; t2 < 48; t2++) a -= Ls[t2 * 48 + i] * y[t2];
      y[i] = a / Ls[i * 48 + i];
    }
    for (int i = 0; i < 48; i++) Wb[tid * 48 + i] = y[i];
  }
  __syncthreads();

  // T1 = L^-1 -> Wa (column tid)
  if (tid < 48) {
    float x[48];
    for (int i = 0; i < 48; i++) {
      float a = (i == tid) ? 1.f : 0.f;
      for (int t2 = 0; t2 < i; t2++) a -= Ls[i * 48 + t2] * x[t2];
      x[i] = a / Ls[i * 48 + i];
    }
    for (int i = 0; i < 48; i++) Wa[i * 48 + tid] = x[i];
  }
  __syncthreads();

  // Aw <- Aw * Aw
  {
    float x[48], y[48];
    if (tid < 48) for (int i = 0; i < 48; i++) x[i] = Aw[i * 48 + tid];
    __syncthreads();
    if (tid < 48) for (int i = 0; i < 48; i++) {
      float a = 0.f;
      for (int j = 0; j < 48; j++) a += Aw[i * 48 + j] * x[j];
      y[i] = a;
    }
    __syncthreads();
    if (tid < 48) for (int i = 0; i < 48; i++) Aw[i * 48 + tid] = y[i];
  }
  __syncthreads();

  // Wa <- L * (Aw(=A^2) * Wa)
  if (tid < 48) {
    float x[48], y[48];
    for (int i = 0; i < 48; i++) x[i] = Wa[i * 48 + tid];
    for (int i = 0; i < 48; i++) { float a = 0.f; for (int j = 0; j < 48; j++) a += Aw[i * 48 + j] * x[j]; y[i] = a; }
    for (int i = 0; i < 48; i++) { float a = 0.f; for (int j = 0; j <= i; j++) a += Ls[i * 48 + j] * y[j]; x[i] = a; }
    for (int i = 0; i < 48; i++) Wa[i * 48 + tid] = x[i];
  }
  __syncthreads();

  // Cm[d][c] = sum_i Wb[d][i] * Wa[i][c]
  if (tid < 48) {
    float x[48];
    for (int i = 0; i < 48; i++) x[i] = Wa[i * 48 + tid];
    for (int d = 0; d < 64; d++) {
      float a = 0.f;
      for (int i = 0; i < 48; i++) a += Wb[d * 48 + i] * x[i];
      Cm[(size_t)bh * 3072 + d * 48 + tid] = a;
    }
  }
}

// ---------------------------------------------------------------------------
// acc[b,t,h*64+d] (+)= gate_k * Cmat[bh] @ q[b,t,h]   (k==0 overwrites)
// ---------------------------------------------------------------------------
__global__ __launch_bounds__(256) void apply_kernel(const float* __restrict__ Q,
                                                    const float* __restrict__ Cm,
                                                    float* __restrict__ acc,
                                                    const float* __restrict__ gate_alphas,
                                                    int k)
{
  __shared__ float Cs[3072];
  __shared__ float sq[16 * 48];
  int bh = blockIdx.x, b = bh >> 4, h = bh & 15, tid = threadIdx.x;
  for (int i = tid; i < 3072; i += 256) Cs[i] = Cm[(size_t)bh * 3072 + i];
  int t0 = blockIdx.y * 16;
  const float* qb = Q + ((size_t)b * T + t0) * HR + h * R;
  for (int i = tid; i < 768; i += 256) {
    int rr = i / 48, cc = i % 48;
    sq[i] = qb[(size_t)rr * HR + cc];
  }
  __syncthreads();
  float gate = 1.f / (1.f + expf(-gate_alphas[k]));
  int tl = tid >> 4, dl = tid & 15;
  float* ab = acc + ((size_t)b * T + t0 + tl) * HHD + h * HD;
#pragma unroll
  for (int j = 0; j < 4; j++) {
    int d = dl + 16 * j;
    float a = 0.f;
#pragma unroll
    for (int r = 0; r < 48; r++) a += Cs[d * 48 + r] * sq[tl * 48 + r];
    a *= gate;
    if (k == 0) ab[d] = a; else ab[d] += a;
  }
}

// ---------------------------------------------------------------------------
extern "C" void kernel_launch(void* const* d_in, const int* in_sizes, int n_in,
                              void* d_out, int out_size, void* d_ws, size_t ws_size,
                              hipStream_t stream)
{
  (void)in_sizes; (void)n_in; (void)out_size; (void)ws_size;
  const float* hs  = (const float*)d_in[0];
  const float* WK  = (const float*)d_in[1];
  const float* WQ  = (const float*)d_in[2];
  const float* WV  = (const float*)d_in[3];
  const float* WO  = (const float*)d_in[4];
  const float* gam = (const float*)d_in[5];
  const float* bet = (const float*)d_in[6];
  const float* gate_alphas = (const float*)d_in[7];
  const float* gate_alpha  = (const float*)d_in[8];
  const float* log_ridges  = (const float*)d_in[9];
  const float* log_gammas  = (const float*)d_in[10];
  const int*   prefix      = (const int*)d_in[11];
  float* out = (float*)d_out;

  char* ws = (char*)d_ws;
  size_t off = 0;
  auto take = [&](size_t bytes) -> char* {
    char* p = ws + off;
    off = (off + bytes + 255) & ~(size_t)255;
    return p;
  };
  __bf16* normedb = (__bf16*)take((size_t)BT * Dm * 2);
  __bf16* WVt     = (__bf16*)take((size_t)Dm * HHD * 2);
  __bf16* WOt     = (__bf16*)take((size_t)HHD * Dm * 2);
  __bf16* WKt     = (__bf16*)take((size_t)KOPS * HR * Dm * 2);
  __bf16* WQt     = (__bf16*)take((size_t)KOPS * HR * Dm * 2);
  float* values   = (float*)take((size_t)BT * HHD * 4);
  float* keys     = (float*)take((size_t)BT * HR * 4);
  float* queries  = (float*)take((size_t)BT * HR * 4);
  float* acc      = (float*)take((size_t)BT * HHD * 4);
  __bf16* accb    = (__bf16*)take((size_t)BT * HHD * 2);
  float* Gb       = (float*)take((size_t)64 * 2304 * 4);
  float* Mb       = (float*)take((size_t)64 * 2304 * 4);
  float* Cvb      = (float*)take((size_t)64 * 3072 * 4);
  float* Cmb      = (float*)take((size_t)64 * 3072 * 4);

  // LayerNorm -> bf16 activations
  ln_kernel<<<BT, 256, 0, stream>>>(hs, gam, bet, normedb);

  // weights -> bf16, transposed (N x K)
  {
    int n = Dm * HHD;
    transpose_bf16_kernel<<<(n + 255) / 256, 256, 0, stream>>>(WV, WVt, Dm, HHD);
    transpose_bf16_kernel<<<(n + 255) / 256, 256, 0, stream>>>(WO, WOt, HHD, Dm);
    int nk = Dm * HR;
    for (int k = 0; k < KOPS; k++) {
      transpose_bf16_kernel<<<(nk + 255) / 256, 256, 0, stream>>>(
          WK + (size_t)k * Dm * HR, WKt + (size_t)k * HR * Dm, Dm, HR);
      transpose_bf16_kernel<<<(nk + 255) / 256, 256, 0, stream>>>(
          WQ + (size_t)k * Dm * HR, WQt + (size_t)k * HR * Dm, Dm, HR);
    }
  }

  // values = normed @ W_V
  gemm_bf16_wmma<<<dim3(BT / 128, HHD / 64), 256, 0, stream>>>(
      normedb, WVt, values, BT, HHD, Dm, nullptr, 0);

  for (int k = 0; k < KOPS; k++) {
    gemm_bf16_wmma<<<dim3(BT / 128, HR / 64), 256, 0, stream>>>(
        normedb, WKt + (size_t)k * HR * Dm, keys, BT, HR, Dm, nullptr, 0);
    gemm_bf16_wmma<<<dim3(BT / 128, HR / 64), 256, 0, stream>>>(
        normedb, WQt + (size_t)k * HR * Dm, queries, BT, HR, Dm, nullptr, 0);
    gm_kernel<<<64, 256, 0, stream>>>(keys, Gb, Mb, prefix);
    cv_kernel<<<64, 256, 0, stream>>>(keys, values, Cvb, prefix);
    factor_kernel<<<64, 64, 0, stream>>>(Gb, Mb, Cvb, Cmb, log_ridges, log_gammas, k);
    apply_kernel<<<dim3(64, T / 16), 256, 0, stream>>>(queries, Cmb, acc, gate_alphas, k);
  }

  // out = sigmoid(gate_alpha) * (acc @ W_O)
  {
    int n = BT * HHD;
    cvt_bf16_kernel<<<(n + 255) / 256, 256, 0, stream>>>(acc, accb, n);
  }
  gemm_bf16_wmma<<<dim3(BT / 128, Dm / 64), 256, 0, stream>>>(
      accb, WOt, out, BT, Dm, HHD, gate_alpha, 1);
}
